// myGraphExpert_72353019069087
// MI455X (gfx1250) — matrix operand order repaired and checked by reading, not compiled, not run on hardware
//
#include <hip/hip_runtime.h>
#include <math.h>

// MI455X / gfx1250, wave32. Flash-attention style fused kernel using
// v_wmma_f32_16x16x32_bf16 for all three matmul stages.

typedef unsigned short u16;
typedef unsigned int   u32;
typedef __attribute__((ext_vector_type(16))) __bf16 v16bf;
typedef __attribute__((ext_vector_type(8)))  float  v8f;
typedef __attribute__((ext_vector_type(4)))  u32    u32x4;

#define WMMA_BF16(a, b, c) \
  __builtin_amdgcn_wmma_f32_16x16x32_bf16(false, (a), false, (b), (short)0, (c), false, false)

// Native bf16 convert (single v_cvt on gfx1250) instead of manual bit-RNE.
__device__ __forceinline__ u16 f2bf(float f) {
  union { __bf16 h; u16 u; } c;
  c.h = (__bf16)f;
  return c.u;
}

union FragU {
  u32x4  q[2];
  v16bf  v;
  u16    s[16];
  __bf16 h[16];
};

// A-fragment (16x32 bf16): lane holds row m=lane&15; K split per half-wave at
// granularity 16 -> one contiguous 32B chunk at (kbase + hl*16).
__device__ __forceinline__ v16bf ldA_bf(const u16* p) {
  FragU f;
  f.q[0] = *(const u32x4*)(p);
  f.q[1] = *(const u32x4*)(p + 8);
  return f.v;
}
// B-fragment (32x16 bf16): lane holds col n=lane&15; K split per half-wave at
// granularity 16 -> one contiguous 32B chunk at (kbase + hl*16).
__device__ __forceinline__ v16bf ldB_bf(const u16* p) {
  FragU f;
  f.q[0] = *(const u32x4*)(p);
  f.q[1] = *(const u32x4*)(p + 8);
  return f.v;
}
// A-fragment gathered from f32 memory with on-the-fly bf16 convert.
__device__ __forceinline__ v16bf ldA_f32(const float* p) {
  FragU f;
#pragma unroll
  for (int i = 0; i < 8; ++i) f.h[i] = (__bf16)p[i];
#pragma unroll
  for (int i = 0; i < 8; ++i) f.h[8 + i] = (__bf16)p[8 + i];
  return f.v;
}

__device__ __forceinline__ v8f vzero8() {
  v8f z;
#pragma unroll
  for (int i = 0; i < 8; ++i) z[i] = 0.0f;
  return z;
}

// XOR-butterfly reductions across the 16-lane group via ds_swizzle
// (offset = xor<<10 | or<<5 | and ; and=0x1f).
#define SWZ(x, imm) __int_as_float(__builtin_amdgcn_ds_swizzle(__float_as_int(x), (imm)))
__device__ __forceinline__ float redmax16(float x) {
  x = fmaxf(x, SWZ(x, 0x041F));
  x = fmaxf(x, SWZ(x, 0x081F));
  x = fmaxf(x, SWZ(x, 0x101F));
  x = fmaxf(x, SWZ(x, 0x201F));
  return x;
}
__device__ __forceinline__ float redsum16(float x) {
  x += SWZ(x, 0x041F);
  x += SWZ(x, 0x081F);
  x += SWZ(x, 0x101F);
  x += SWZ(x, 0x201F);
  return x;
}

// ---------------------------------------------------------------------------
// Kernel 1: W1/W2 f32 -> bf16 (16384 elements each)
__global__ void k_wconv_72353019069087(const float* __restrict__ W1,
                                       const float* __restrict__ W2,
                                       u16* __restrict__ wb1, u16* __restrict__ wb2) {
  int i = blockIdx.x * blockDim.x + threadIdx.x;  // exactly 16384 threads
  wb1[i] = f2bf(W1[i]);
  wb2[i] = f2bf(W2[i]);
}

// ---------------------------------------------------------------------------
// Kernel 2: k = x @ W2^T + b2 (bf16 out), plus x^T in bf16 per (b,h).
// One wave per 16-row tile of the flattened [B*H*L, 128] matrix.
__global__ __launch_bounds__(128) void k_proj_72353019069087(
    const float* __restrict__ x, const u16* __restrict__ wb2,
    const float* __restrict__ b2, u16* __restrict__ kb, u16* __restrict__ xT) {
  const int lane = threadIdx.x & 31;
  const int wave = threadIdx.x >> 5;
  const int l15 = lane & 15, hl = lane >> 4;
  const int w = blockIdx.x * 4 + wave;  // 0..8191
  const int row0 = w << 4;
  const int bh = row0 >> 11;
  const int l0 = row0 & 2047;

  v16bf ax[4];
#pragma unroll
  for (int kk = 0; kk < 4; ++kk)
    ax[kk] = ldA_f32(x + (size_t)(row0 + l15) * 128 + kk * 32 + hl * 16);

#pragma unroll
  for (int t = 0; t < 8; ++t) {
    v8f acc = vzero8();
#pragma unroll
    for (int kk = 0; kk < 4; ++kk) {
      v16bf bw = ldB_bf(wb2 + (size_t)(t * 16 + l15) * 128 + kk * 32 + hl * 16);
      acc = WMMA_BF16(ax[kk], bw, acc);
    }
    float bias = b2[t * 16 + l15];
#pragma unroll
    for (int v = 0; v < 8; ++v)
      kb[(size_t)(row0 + hl * 8 + v) * 128 + t * 16 + l15] = f2bf(acc[v] + bias);
  }

  // Write x^T (bf16): xT[bh][d][l], packed 2 bf16 per dword store.
  u16* xTb = xT + (size_t)bh * 128 * 2048;
  for (int it = 0; it < 32; ++it) {
    int idx = it * 32 + lane;          // 0..1023  -> (d, row-pair)
    int d = idx >> 3, p2 = (idx & 7) << 1;
    float a0 = x[(size_t)(row0 + p2) * 128 + d];
    float a1 = x[(size_t)(row0 + p2 + 1) * 128 + d];
    u32 pk = (u32)f2bf(a0) | ((u32)f2bf(a1) << 16);
    *(u32*)(xTb + (size_t)d * 2048 + l0 + p2) = pk;
  }
}

// ---------------------------------------------------------------------------
// Kernel 3: fused q-projection + GELU(QK^T) + online softmax + P@X.
// One wave per 16-query tile; j-loop over keys in steps of 32.
__global__ __launch_bounds__(128) void k_attn_72353019069087(
    const float* __restrict__ x, const u16* __restrict__ wb1,
    const float* __restrict__ b1, const u16* __restrict__ kb,
    const u16* __restrict__ xT, float* __restrict__ out) {
  __shared__ __align__(16) u16 ldsQ[4][16 * 128];
  __shared__ __align__(16) u16 ldsP[4][16 * 32];

  const int lane = threadIdx.x & 31;
  const int wave = threadIdx.x >> 5;
  const int l15 = lane & 15, hl = lane >> 4;
  const int w = blockIdx.x * 4 + wave;  // 0..8191
  const int bh = w >> 7;
  const int i0 = (w & 127) << 4;

  const float* xb  = x  + (size_t)bh * 2048 * 128;
  const u16*   kbh = kb + (size_t)bh * 2048 * 128;
  const u16*   xTb = xT + (size_t)bh * 128 * 2048;
  u16* lq = ldsQ[wave];
  u16* lp = ldsP[wave];

  // ---- q = x @ W1^T + b1 for this 16-row tile; stage bf16 q in LDS so we
  //      can re-read it in A-fragment layout (D-layout -> A-layout transpose).
  {
    v16bf ax[4];
#pragma unroll
    for (int kk = 0; kk < 4; ++kk)
      ax[kk] = ldA_f32(xb + (size_t)(i0 + l15) * 128 + kk * 32 + hl * 16);
#pragma unroll
    for (int t = 0; t < 8; ++t) {
      v8f qa = vzero8();
#pragma unroll
      for (int kk = 0; kk < 4; ++kk) {
        v16bf bw = ldB_bf(wb1 + (size_t)(t * 16 + l15) * 128 + kk * 32 + hl * 16);
        qa = WMMA_BF16(ax[kk], bw, qa);
      }
      float bias = b1[t * 16 + l15];
#pragma unroll
      for (int v = 0; v < 8; ++v)
        lq[(hl * 8 + v) * 128 + t * 16 + l15] = f2bf(qa[v] + bias);
    }
  }
  __syncthreads();

  v16bf aq[4];
#pragma unroll
  for (int kk = 0; kk < 4; ++kk)
    aq[kk] = ldA_bf(lq + l15 * 128 + kk * 32 + hl * 16);

  v8f acc[8];
  float rmax[8], rsum[8];
#pragma unroll
  for (int t = 0; t < 8; ++t) acc[t] = vzero8();
#pragma unroll
  for (int v = 0; v < 8; ++v) { rmax[v] = -3.0e38f; rsum[v] = 0.0f; }

  for (int j0 = 0; j0 < 2048; j0 += 32) {
    // Prefetch next iteration's key rows toward the WGP (global_prefetch_b8).
    if (j0 + 32 < 2048) {
      const u16* nk = kbh + (size_t)(j0 + 32 + l15) * 128 + hl * 16;
      __builtin_prefetch(nk, 0, 1);
      __builtin_prefetch(nk + 16 * 128, 0, 1);
    }
    // S = q @ k^T for 32 keys (two 16x16 score fragments)
    v8f s0 = vzero8(), s1 = vzero8();
#pragma unroll
    for (int kk = 0; kk < 4; ++kk) {
      v16bf bk0 = ldB_bf(kbh + (size_t)(j0 + l15) * 128 + kk * 32 + hl * 16);
      v16bf bk1 = ldB_bf(kbh + (size_t)(j0 + 16 + l15) * 128 + kk * 32 + hl * 16);
      s0 = WMMA_BF16(aq[kk], bk0, s0);
      s1 = WMMA_BF16(aq[kk], bk1, s1);
    }
    // exact-erf GELU on raw scores, then online softmax bookkeeping
    float sc[8];
#pragma unroll
    for (int v = 0; v < 8; ++v) {
      float g0 = s0[v], g1 = s1[v];
      g0 = 0.5f * g0 * (1.0f + erff(g0 * 0.70710678118654752f));
      g1 = 0.5f * g1 * (1.0f + erff(g1 * 0.70710678118654752f));
      float tm = redmax16(fmaxf(g0, g1));
      float nm = fmaxf(rmax[v], tm);
      sc[v] = __expf(rmax[v] - nm);
      rmax[v] = nm;
      float p0 = __expf(g0 - nm);
      float p1 = __expf(g1 - nm);
      rsum[v] = rsum[v] * sc[v] + redsum16(p0 + p1);
      // stage P (bf16) row-major 16x32 for A-fragment re-read
      lp[(hl * 8 + v) * 32 + l15]      = f2bf(p0);
      lp[(hl * 8 + v) * 32 + 16 + l15] = f2bf(p1);
    }
#pragma unroll
    for (int t = 0; t < 8; ++t) {
#pragma unroll
      for (int v = 0; v < 8; ++v) acc[t][v] *= sc[v];
    }
    v16bf ap = ldA_bf(lp + l15 * 32 + hl * 16);
    // acc += P @ X  (B-operand from x^T, row-contiguous)
#pragma unroll
    for (int t = 0; t < 8; ++t) {
      v16bf bx = ldB_bf(xTb + (size_t)(t * 16 + l15) * 2048 + j0 + hl * 16);
      acc[t] = WMMA_BF16(ap, bx, acc[t]);
    }
  }

  // epilogue: divide by row sum, scatter to out[b, l, h*128 + d]
  const int bb = bh >> 4, hh = bh & 15;
#pragma unroll
  for (int v = 0; v < 8; ++v) {
    float inv = 1.0f / rsum[v];
    float* orow = out + ((size_t)(bb * 2048 + i0 + hl * 8 + v)) * 2048 + hh * 128;
#pragma unroll
    for (int t = 0; t < 8; ++t) orow[t * 16 + l15] = acc[t][v] * inv;
  }
}

// ---------------------------------------------------------------------------
extern "C" void kernel_launch(void* const* d_in, const int* in_sizes, int n_in,
                              void* d_out, int out_size, void* d_ws, size_t ws_size,
                              hipStream_t stream) {
  (void)in_sizes; (void)n_in; (void)out_size; (void)ws_size;
  const float* x  = (const float*)d_in[0];
  const float* W1 = (const float*)d_in[1];
  const float* b1 = (const float*)d_in[2];
  const float* W2 = (const float*)d_in[3];
  const float* b2 = (const float*)d_in[4];
  float* out = (float*)d_out;

  // workspace layout (bf16): W1 (32KB) | W2 (32KB) | k (32MB) | xT (32MB)
  u16* wb1 = (u16*)d_ws;
  u16* wb2 = wb1 + 128 * 128;
  u16* kbw = wb2 + 128 * 128;
  u16* xT  = kbw + (size_t)131072 * 128;

  k_wconv_72353019069087<<<64, 256, 0, stream>>>(W1, W2, wb1, wb2);
  k_proj_72353019069087<<<2048, 128, 0, stream>>>(x, wb2, b2, kbw, xT);
  k_attn_72353019069087<<<2048, 128, 0, stream>>>(x, wb1, b1, kbw, xT, out);
}
